// UnifiedParisiNashAttention_38027640439251
// MI455X (gfx1250) — matrix operand order repaired and checked by reading, not compiled, not run on hardware
//
#include <hip/hip_runtime.h>

// ---------------------------------------------------------------------------
// MoE block: router(top-1) + per-expert sliding-window attention + SwiGLU.
// Heavy GEMMs via v_wmma_f32_16x16x32_bf16, LDS double-buffered, fragments
// fetched as ds_load_b128 from pre-packed LDS layouts.
// ---------------------------------------------------------------------------

typedef __attribute__((ext_vector_type(8)))  float  v8f;
typedef __attribute__((ext_vector_type(16))) __bf16 v16bf;
typedef __attribute__((ext_vector_type(2)))  __bf16 v2bf;

#define B_    2
#define L_    1024
#define D_    768
#define F_    3072
#define E_    8
#define H_    12
#define HD_   64
#define WIN_  128
#define NTOK  (B_ * L_)

#define ASTR  20   // LDS row stride (dwords) for packed A rows (16 pairs + pad)
#define BSTR  20   // LDS col stride (dwords) for packed B cols (16 pairs + pad)

__device__ __forceinline__ unsigned int pack2bf(float lo, float hi) {
#if __has_builtin(__builtin_amdgcn_cvt_pk_bf16_f32)
    union { v2bf v; unsigned int u; } r;
    r.v = __builtin_amdgcn_cvt_pk_bf16_f32(lo, hi);   // single v_cvt_pk_bf16_f32
    return r.u;
#else
    // round-half-up to bf16, then one v_perm_b32 merges the two high halves:
    // dst = { hi.b3, hi.b2, lo.b3, lo.b2 }
    unsigned int ul = __float_as_uint(lo) + 0x8000u;
    unsigned int uh = __float_as_uint(hi) + 0x8000u;
    return __builtin_amdgcn_perm(uh, ul, 0x07060302u);
#endif
}

// ---------------------------------------------------------------------------
// WMMA GEMM: C[M,N] = A[M,K] (f32 rm) @ Bw[K,N] (f32 rm), bf16 math, f32 acc.
// Block tile 128x128, K-step 32, 256 threads = 8 waves (4 row x 2 col).
// Each wave: 32x64 of C = 2 row strips x 4 col tiles = 8 WMMAs / K-step.
// LDS packed layouts (K-pairs per dword):
//   sA[buf][row*ASTR + kp]  kp = K/2 in 0..15  -> lane fragment = 2x b128
//   sB[buf][col*BSTR + kp]                      -> lane fragment = 2x b128
// MODE 0: plain store. MODE 1: store iff idx[row]==expert.
// MODE 2: C = silu(gbuf) * acc  (fused SwiGLU gate).
// ---------------------------------------------------------------------------
template <int MODE>
__global__ __launch_bounds__(256)
void gemm_bf16(const float* __restrict__ A, const float* __restrict__ Bw,
               float* __restrict__ C, int M, int N, int K,
               const int* __restrict__ idx, int expert,
               const float* __restrict__ gbuf)
{
    (void)M;
    __shared__ __align__(16) unsigned int sA[2][128 * ASTR];  // 2 x 10 KB
    __shared__ __align__(16) unsigned int sB[2][128 * BSTR];  // 2 x 10 KB

    const int tid  = threadIdx.x;
    const int lane = tid & 31;
    const int w    = tid >> 5;
    const int wm   = w & 3;          // row strip group: 32 rows
    const int wn   = w >> 2;         // col half: 64 cols
    const int brow = blockIdx.y * 128;
    const int bcol = blockIdx.x * 128;
    const int m    = lane & 15;
    const int hh   = lane >> 4;

    v8f acc[2][4];
    #pragma unroll
    for (int rs = 0; rs < 2; ++rs)
        #pragma unroll
        for (int nt = 0; nt < 4; ++nt)
            #pragma unroll
            for (int j = 0; j < 8; ++j) acc[rs][nt][j] = 0.0f;

    // staging registers for next tile
    float2 areg[8];
    float  b0reg[8], b1reg[8];
    const int bc  = tid & 127;         // B staging column
    const int brg = (tid >> 7) * 8;    // B staging pair-row group

    auto load_tile = [&](int k0) {
        #pragma unroll
        for (int j = 0; j < 8; ++j) {      // A: 128 rows x 16 pairs / 256 thr
            int e = tid + j * 256;
            int r = e >> 4, kp = e & 15;
            areg[j] = *(const float2*)&A[(size_t)(brow + r) * K + k0 + 2 * kp];
        }
        #pragma unroll
        for (int j = 0; j < 8; ++j) {      // B: 16 pair-rows x 128 cols
            int r2 = brg + j;
            b0reg[j] = Bw[(size_t)(k0 + 2 * r2) * N + bcol + bc];
            b1reg[j] = Bw[(size_t)(k0 + 2 * r2 + 1) * N + bcol + bc];
        }
    };
    auto store_tile = [&](int buf) {
        #pragma unroll
        for (int j = 0; j < 8; ++j) {
            int e = tid + j * 256;
            int r = e >> 4, kp = e & 15;
            sA[buf][r * ASTR + kp] = pack2bf(areg[j].x, areg[j].y);
        }
        #pragma unroll
        for (int j = 0; j < 8; ++j)
            sB[buf][bc * BSTR + brg + j] = pack2bf(b0reg[j], b1reg[j]);
    };

    load_tile(0);
    store_tile(0);
    __syncthreads();

    const int nk = K >> 5;
    for (int t = 0; t < nk; ++t) {
        const int buf = t & 1;
        if (t + 1 < nk) load_tile((t + 1) * 32);     // issue global loads early
        if (t + 2 < nk) {                            // global_prefetch_b8 t+2
            __builtin_prefetch(&A[(size_t)(brow + (tid >> 1)) * K + (t + 2) * 32], 0, 1);
            __builtin_prefetch(&Bw[(size_t)((t + 2) * 32 + (tid >> 3)) * N + bcol + (tid & 7) * 16], 0, 1);
        }

        // A fragments: two row strips, each 2x ds_load_b128
        union Frag { uint4 q[2]; unsigned int u[8]; v16bf v; };
        Frag af[2];
        #pragma unroll
        for (int rs = 0; rs < 2; ++rs) {
            const int row = wm * 32 + rs * 16 + m;
            af[rs].q[0] = *(const uint4*)&sA[buf][row * ASTR + hh * 4];
            af[rs].q[1] = *(const uint4*)&sA[buf][row * ASTR + 8 + hh * 4];
        }
        #pragma unroll
        for (int nt = 0; nt < 4; ++nt) {
            Frag bfr;
            const int c = wn * 64 + nt * 16 + m;
            bfr.q[0] = *(const uint4*)&sB[buf][c * BSTR + hh * 4];
            bfr.q[1] = *(const uint4*)&sB[buf][c * BSTR + 8 + hh * 4];
            acc[0][nt] = __builtin_amdgcn_wmma_f32_16x16x32_bf16(
                false, af[0].v, false, bfr.v, (short)0, acc[0][nt], false, false);
            acc[1][nt] = __builtin_amdgcn_wmma_f32_16x16x32_bf16(
                false, af[1].v, false, bfr.v, (short)0, acc[1][nt], false, false);
        }

        if (t + 1 < nk) store_tile(buf ^ 1);         // convert + stage next
        __syncthreads();
    }

    // Epilogue: lane col = m, VGPR v row = v + 8*hh within a 16-row tile
    #pragma unroll
    for (int rs = 0; rs < 2; ++rs) {
        #pragma unroll
        for (int nt = 0; nt < 4; ++nt) {
            #pragma unroll
            for (int v = 0; v < 8; ++v) {
                int row = brow + wm * 32 + rs * 16 + v + 8 * hh;
                int col = bcol + wn * 64 + nt * 16 + m;
                float val = acc[rs][nt][v];
                if (MODE == 0) {
                    C[(size_t)row * N + col] = val;
                } else if (MODE == 1) {
                    if (idx[row] == expert) C[(size_t)row * N + col] = val;
                } else {  // MODE 2: fused SwiGLU gate
                    float g = gbuf[(size_t)row * N + col];
                    C[(size_t)row * N + col] = (g / (1.0f + __expf(-g))) * val;
                }
            }
        }
    }
}

// ---------------------------------------------------------------------------
// Router: per-token logits = (x @ gate_w)/2, softmax, argmax (first max).
// ---------------------------------------------------------------------------
__global__ void router_kernel(const float* __restrict__ x,
                              const float* __restrict__ gw,
                              int* __restrict__ idx,
                              float* __restrict__ PT)
{
    int t = blockIdx.x * blockDim.x + threadIdx.x;
    if (t >= NTOK) return;
    float acc[E_];
    #pragma unroll
    for (int e = 0; e < E_; ++e) acc[e] = 0.0f;
    const float* xr = x + (size_t)t * D_;
    for (int d = 0; d < D_; ++d) {
        float xv = xr[d];
        #pragma unroll
        for (int e = 0; e < E_; ++e) acc[e] += xv * gw[d * E_ + e];
    }
    float mx = -3.4e38f; int am = 0;
    #pragma unroll
    for (int e = 0; e < E_; ++e) {
        acc[e] *= 0.5f;                               // 1/TEMP
        if (acc[e] > mx) { mx = acc[e]; am = e; }     // strict > keeps first
    }
    float s = 0.0f, p[E_];
    #pragma unroll
    for (int e = 0; e < E_; ++e) { p[e] = __expf(acc[e] - mx); s += p[e]; }
    float inv = 1.0f / s;
    idx[t] = am;
    #pragma unroll
    for (int e = 0; e < E_; ++e) PT[t * E_ + e] = p[e] * inv;
}

// Deterministic aux-loss: one wave; fixed-tree shfl reductions.
__global__ void aux_kernel(const float* __restrict__ PT,
                           const int* __restrict__ IDX,
                           float* __restrict__ out_aux)
{
    const int lane = threadIdx.x;
    float aux = 0.0f;
    for (int e = 0; e < E_; ++e) {
        float ps = 0.0f, fs = 0.0f;
        for (int t = lane; t < NTOK; t += 32) {
            ps += PT[t * E_ + e];
            fs += (IDX[t] == e) ? 1.0f : 0.0f;
        }
        #pragma unroll
        for (int mm = 16; mm >= 1; mm >>= 1) {
            ps += __shfl_xor(ps, mm, 32);
            fs += __shfl_xor(fs, mm, 32);
        }
        aux += (fs / (float)NTOK) * (ps / (float)NTOK);
    }
    if (lane == 0) *out_aux = (float)E_ * aux;
}

// Column sums of K over L (cohesion mean is linear: mu_q = q . Ksum / (8*L)).
__global__ void ksum_kernel(const float* __restrict__ Kb, float* __restrict__ ks)
{
    int g = blockIdx.x * blockDim.x + threadIdx.x;
    if (g >= B_ * D_) return;
    int b = g / D_, c = g % D_;
    float s = 0.0f;
    const float* p = Kb + (size_t)b * L_ * D_ + c;
    for (int l = 0; l < L_; ++l) s += p[(size_t)l * D_];
    ks[g] = s;
}

// ---------------------------------------------------------------------------
// Sliding-window attention with reynolds modulation. One wave per (b,h,q).
// ---------------------------------------------------------------------------
__global__ __launch_bounds__(256)
void attn_kernel(const float* __restrict__ Q, const float* __restrict__ Kb,
                 const float* __restrict__ Vb, const float* __restrict__ ks,
                 float* __restrict__ AO)
{
    const int lane = threadIdx.x & 31;
    const int gw   = blockIdx.x * 8 + (threadIdx.x >> 5);
    const int q = gw % L_;
    const int h = (gw / L_) % H_;
    const int b = gw / (L_ * H_);
    const int d0 = lane * 2;
    const size_t qoff = (size_t)(b * L_ + q) * D_ + h * HD_;
    const float qa = Q[qoff + d0], qb = Q[qoff + d0 + 1];

    float part = qa * ks[b * D_ + h * HD_ + d0] + qb * ks[b * D_ + h * HD_ + d0 + 1];
    #pragma unroll
    for (int mm = 16; mm >= 1; mm >>= 1) part += __shfl_xor(part, mm, 32);
    const float mu = part * (1.0f / (8.0f * (float)L_));

    const int k0 = (q >= WIN_ - 1) ? (q - (WIN_ - 1)) : 0;
    float mrun = -3.4e38f, lsum = 0.0f, oa = 0.0f, ob = 0.0f;
    for (int k = k0; k <= q; ++k) {
        const size_t koff = (size_t)(b * L_ + k) * D_ + h * HD_;
        float p = qa * Kb[koff + d0] + qb * Kb[koff + d0 + 1];
        #pragma unroll
        for (int mm = 16; mm >= 1; mm >>= 1) p += __shfl_xor(p, mm, 32);
        float s = p * 0.125f;
        float sim = 1.0f / (1.0f + __expf(-s));
        s = s - 0.1f * sim * sim + 0.1f * s - 0.1f * fabsf(s - mu);
        float nm    = fmaxf(mrun, s);
        float scale = __expf(mrun - nm);
        float wgt   = __expf(s - nm);
        lsum = lsum * scale + wgt;
        oa   = oa * scale + wgt * Vb[koff + d0];
        ob   = ob * scale + wgt * Vb[koff + d0 + 1];
        mrun = nm;
    }
    const float invl = 1.0f / lsum;
    AO[qoff + d0]     = oa * invl;
    AO[qoff + d0 + 1] = ob * invl;
}

// ---------------------------------------------------------------------------
extern "C" void kernel_launch(void* const* d_in, const int* in_sizes, int n_in,
                              void* d_out, int out_size, void* d_ws, size_t ws_size,
                              hipStream_t stream)
{
    (void)in_sizes; (void)n_in; (void)out_size; (void)ws_size;
    const float* x     = (const float*)d_in[0];
    const float* gate  = (const float*)d_in[1];
    const float* q_w   = (const float*)d_in[2];
    const float* k_w   = (const float*)d_in[3];
    const float* v_w   = (const float*)d_in[4];
    const float* w1    = (const float*)d_in[5];
    const float* vg    = (const float*)d_in[6];
    const float* w2    = (const float*)d_in[7];
    const float* out_w = (const float*)d_in[8];
    float* out = (float*)d_out;

    char* ws = (char*)d_ws;
    float* Qb  = (float*)ws;  ws += (size_t)NTOK * D_ * 4;
    float* Kb  = (float*)ws;  ws += (size_t)NTOK * D_ * 4;
    float* Vb  = (float*)ws;  ws += (size_t)NTOK * D_ * 4;
    float* AO  = (float*)ws;  ws += (size_t)NTOK * D_ * 4;
    float* SEL = (float*)ws;  ws += (size_t)NTOK * D_ * 4;
    float* G1  = (float*)ws;  ws += (size_t)NTOK * F_ * 4;
    float* KS  = (float*)ws;  ws += (size_t)B_ * D_ * 4;
    float* PT  = (float*)ws;  ws += (size_t)NTOK * E_ * 4;
    int*   IDX = (int*)ws;    ws += (size_t)NTOK * 4;

    router_kernel<<<NTOK / 256, 256, 0, stream>>>(x, gate, IDX, PT);
    aux_kernel<<<1, 32, 0, stream>>>(PT, IDX, out + (size_t)NTOK * D_);

    dim3 gD(D_ / 128, NTOK / 128);   // N = 768
    dim3 gF(F_ / 128, NTOK / 128);   // N = 3072
    const size_t dd = (size_t)D_ * D_;
    const size_t df = (size_t)D_ * F_;
    const size_t fd = (size_t)F_ * D_;

    for (int e = 0; e < E_; ++e) {
        gemm_bf16<0><<<gD, 256, 0, stream>>>(x, q_w + e * dd, Qb, NTOK, D_, D_, nullptr, 0, nullptr);
        gemm_bf16<0><<<gD, 256, 0, stream>>>(x, k_w + e * dd, Kb, NTOK, D_, D_, nullptr, 0, nullptr);
        gemm_bf16<0><<<gD, 256, 0, stream>>>(x, v_w + e * dd, Vb, NTOK, D_, D_, nullptr, 0, nullptr);
        ksum_kernel<<<(B_ * D_ + 255) / 256, 256, 0, stream>>>(Kb, KS);
        attn_kernel<<<B_ * H_ * L_ / 8, 256, 0, stream>>>(Qb, Kb, Vb, KS, AO);
        gemm_bf16<0><<<gF, 256, 0, stream>>>(AO, w1 + e * df, G1, NTOK, F_, D_, nullptr, 0, nullptr);
        // vg GEMM with fused SiLU gate: G1 <- silu(G1) * (AO @ vg)
        gemm_bf16<2><<<gF, 256, 0, stream>>>(AO, vg + e * df, G1, NTOK, F_, D_, nullptr, 0, G1);
        // w2 GEMM with fused top-1 selection in the epilogue
        gemm_bf16<1><<<gD, 256, 0, stream>>>(G1, w2 + e * fd, SEL, NTOK, D_, F_, IDX, e, nullptr);
    }
    gemm_bf16<0><<<gD, 256, 0, stream>>>(SEL, out_w, out, NTOK, D_, D_, nullptr, 0, nullptr);
}
